// MultiHeadAttention_67989332296191
// MI455X (gfx1250) — compile-verified, matrix operand order checked by reading, not run on hardware
//
#include <hip/hip_runtime.h>

typedef __attribute__((ext_vector_type(16))) __bf16 v16bf;
typedef __attribute__((ext_vector_type(8)))  float  v8f;
typedef __attribute__((ext_vector_type(4)))  unsigned int u32x4;
typedef __attribute__((ext_vector_type(8)))  int    i32x8;
typedef __attribute__((ext_vector_type(4)))  int    i32x4;

#define D_MODEL 1024
#define NHEAD   16
#define DEPTH   64
#define SEQQ    2048
#define SEQP    2048
#define SEQT    4096          // SEQP + SEQQ
#define BATCH   2
#define M_TOT   4096          // BATCH * SEQQ
#define NEG_INF (-1000000000.0f)

// ---- feature probes (report via stderr; compile-only ground truth) --------
#if defined(__HIP_DEVICE_COMPILE__)
#if __has_builtin(__builtin_amdgcn_tensor_load_to_lds)
#define USE_TDM 1
#pragma message("CDNA5 probe: tensor_load_to_lds builtin PRESENT -> TDM path enabled")
#else
#define USE_TDM 0
#pragma message("CDNA5 probe: tensor_load_to_lds builtin ABSENT -> manual Q staging")
#endif
#if __has_builtin(__builtin_amdgcn_global_load_async_to_lds_b128)
#pragma message("CDNA5 probe: global_load_async_to_lds_b128 builtin PRESENT")
#endif
#if __has_builtin(__builtin_amdgcn_load_to_lds)
#pragma message("CDNA5 probe: load_to_lds builtin PRESENT")
#endif
#if __has_builtin(__builtin_amdgcn_global_load_lds)
#pragma message("CDNA5 probe: global_load_lds builtin PRESENT")
#endif
#else
#define USE_TDM 0
#endif

static __device__ inline v8f wmma_bf16(v16bf a, v16bf b, v8f c) {
    // D = A(16x32 bf16) * B(32x16 bf16) + C(16x16 f32)
    return __builtin_amdgcn_wmma_f32_16x16x32_bf16(
        /*neg_a=*/false, a, /*neg_b=*/false, b,
        /*c_mod=*/(short)0, c, /*reuse_a=*/false, /*reuse_b=*/false);
}

// ---------------------------------------------------------------------------
// Kernel A: C = X @ W1 + b1  (M=4096, K=1024, N=3072), scatter q/k/v.
// One wave per 64x16 output tile (4 sub-tiles reuse one W1 B-frag).
// A-frag layout (16-bit A 16x32): lane<16 -> M=lane, K in {0..7,16..23};
// lane>=16 -> M=lane-16, K in {8..15,24..31}. B-frag mirrored (lane&15 = N).
// ---------------------------------------------------------------------------
__global__ __launch_bounds__(32)
void qkv_kernel(const float* __restrict__ X, const float* __restrict__ W1,
                const float* __restrict__ b1, float* __restrict__ present,
                __bf16* __restrict__ qws, __bf16* __restrict__ kws,
                __bf16* __restrict__ vws)
{
    const int lane = threadIdx.x & 31;
    const int half = lane >> 4;
    const int l16  = lane & 15;
    const int koff = half * 8;
    const int n0 = blockIdx.x * 16;   // 0..3071
    const int m0 = blockIdx.y * 64;   // 0..4095

    v8f acc[4] = {{}, {}, {}, {}};
    for (int kk = 0; kk < D_MODEL; kk += 32) {
        v16bf b;
        const float* wr = W1 + (size_t)(kk + koff) * (3 * D_MODEL) + n0 + l16;
#pragma unroll
        for (int j = 0; j < 8; ++j) {
            b[j]     = (__bf16)wr[(size_t)j * (3 * D_MODEL)];
            b[8 + j] = (__bf16)wr[(size_t)(16 + j) * (3 * D_MODEL)];
        }
#pragma unroll
        for (int sm = 0; sm < 4; ++sm) {
            v16bf a;
            const float* xr = X + (size_t)(m0 + sm * 16 + l16) * D_MODEL + kk + koff;
#pragma unroll
            for (int j = 0; j < 8; ++j) {
                a[j]     = (__bf16)xr[j];
                a[8 + j] = (__bf16)xr[16 + j];
            }
            acc[sm] = wmma_bf16(a, b, acc[sm]);
        }
    }

    const int   n    = n0 + l16;
    const float bias = b1[n];
    const int   sec  = n >> 10;        // 0=q, 1=k, 2=v (uniform over wave)
    const int   cc   = n & 1023;
    const int   h    = cc >> 6;
    const int   d    = cc & 63;

#pragma unroll
    for (int sm = 0; sm < 4; ++sm) {
#pragma unroll
        for (int g = 0; g < 8; ++g) {
            const int m  = m0 + sm * 16 + g + half * 8;
            const int bb = m >> 11;        // batch
            const int s  = m & 2047;       // seq pos
            const float v = acc[sm][g] + bias;
            if (sec == 0) {
                // q, pre-scaled by 1/sqrt(DEPTH)
                qws[((size_t)(bb * NHEAD + h) * SEQQ + s) * DEPTH + d] =
                    (__bf16)(v * 0.125f);
            } else {
                // present: (B, 2, H, SEQQ, DEPTH), fp32; write-once -> NT store
                size_t pidx = (((size_t)(bb * 2 + (sec - 1)) * NHEAD + h) * SEQQ + s)
                                  * DEPTH + d;
                __builtin_nontemporal_store(v, present + pidx);
                __bf16* dst = (sec == 1) ? kws : vws;
                dst[((size_t)(bb * NHEAD + h) * SEQT + SEQP + s) * DEPTH + d] =
                    (__bf16)v;
            }
        }
    }
}

// ---------------------------------------------------------------------------
// Kernel B: past (B,2,H,SEQP,DEPTH) fp32 -> kws/vws bf16 at positions 0..SEQP
// past is read-once -> NT loads keep it out of L2's reuse set.
// ---------------------------------------------------------------------------
__global__ __launch_bounds__(256)
void past_kernel(const float* __restrict__ past,
                 __bf16* __restrict__ kws, __bf16* __restrict__ vws)
{
    const int idx = blockIdx.x * 256 + threadIdx.x;   // < 2^22
    const int d  = idx & 63;
    const int sp = (idx >> 6) & (SEQP - 1);
    const int h  = (idx >> 17) & 15;
    const int bb = idx >> 21;
    const size_t srck = (((size_t)(bb * 2 + 0) * NHEAD + h) * SEQP + sp) * DEPTH + d;
    const size_t srcv = (((size_t)(bb * 2 + 1) * NHEAD + h) * SEQP + sp) * DEPTH + d;
    const size_t dst  = ((size_t)(bb * NHEAD + h) * SEQT + sp) * DEPTH + d;
    kws[dst] = (__bf16)__builtin_nontemporal_load(past + srck);
    vws[dst] = (__bf16)__builtin_nontemporal_load(past + srcv);
}

// ---------------------------------------------------------------------------
// Kernel C: fused logits + mask + softmax + attn write + P@V.
// One workgroup (8 waves) per (b*h, 16 q rows). LDS: 16x4096 f32 logits tile.
// Q tile staged via the Tensor Data Mover (tensor_load_to_lds + s_wait_tensorcnt).
// ---------------------------------------------------------------------------
__global__ __launch_bounds__(256)
void attn_kernel(const __bf16* __restrict__ qws, const __bf16* __restrict__ kws,
                 const __bf16* __restrict__ vws, const float* __restrict__ mask,
                 float* __restrict__ attn_out, __bf16* __restrict__ saws)
{
    extern __shared__ float smem[];
    float*  rowbuf = smem;                       // 16 * 4096 f32 = 256 KB
    float*  red    = smem + 16 * SEQT;           // 512 f32 (max + sum scratch)
    float*  saacc  = red + 512;                  // 16 * 64 f32 partial-sum tile
    __bf16* qtile  = (__bf16*)(saacc + 16 * 64); // 16 * 64 bf16 Q staging

    const int tid  = threadIdx.x;
    const int wave = tid >> 5;
    const int lane = tid & 31;
    const int half = lane >> 4;
    const int l16  = lane & 15;
    const int koff = half * 8;
    const int bh   = blockIdx.y;           // b*NHEAD + h
    const int q0   = blockIdx.x * 16;

    // --- stage Q tile (16 x 64 bf16) into LDS ---
#if USE_TDM
    if (tid < 32) {
        const unsigned long long gaddr =
            (unsigned long long)(uintptr_t)(qws + ((size_t)bh * SEQQ + q0) * DEPTH);
        const unsigned int lds_off = (unsigned int)(uintptr_t)(void*)qtile;
        u32x4 g0;
        g0[0] = 1u;                                   // count=1, user descriptor
        g0[1] = lds_off;                              // LDS byte address
        g0[2] = (unsigned int)gaddr;                  // global_addr[31:0]
        g0[3] = (unsigned int)((gaddr >> 32) & 0x01FFFFFFu) | (2u << 30); // [56:32] | type=2
        i32x8 g1;
        g1[0] = (int)(1u << 16);                      // data_size = 1 (2 bytes)
        g1[1] = (int)((unsigned)DEPTH << 16);         // tensor_dim0 = 64
        g1[2] = (int)(16u << 16);                     // tensor_dim1 = 16 (rows left)
        g1[3] = (int)((unsigned)DEPTH << 16);         // tile_dim0 = 64
        g1[4] = 16;                                   // tile_dim1 = 16 rows
        g1[5] = DEPTH;                                // tensor_dim0_stride = 64 elems
        g1[6] = 0;
        g1[7] = 0;
        i32x4 z4 = {0, 0, 0, 0};
#if defined(__clang_major__) && (__clang_major__ >= 23)
        i32x8 z8 = {0, 0, 0, 0, 0, 0, 0, 0};
        __builtin_amdgcn_tensor_load_to_lds(g0, g1, z4, z4, z8, 0);
#else
        __builtin_amdgcn_tensor_load_to_lds(g0, g1, z4, z4, 0);
#endif
        __builtin_amdgcn_s_wait_tensorcnt(0);
    }
    __syncthreads();
#else
    if (tid < 32) {
        const __bf16* src = qws + ((size_t)bh * SEQQ + q0) * DEPTH;
        for (int i = lane; i < 16 * DEPTH; i += 32) qtile[i] = src[i];
    }
    __syncthreads();
#endif

    // --- load Q A-frags from LDS (16 rows x 64 depth, already scaled) ---
    v16bf aq0, aq1;
    {
        const __bf16* qr = qtile + (size_t)l16 * DEPTH;
#pragma unroll
        for (int j = 0; j < 8; ++j) {
            aq0[j]     = qr[koff + j];
            aq0[8 + j] = qr[16 + koff + j];
            aq1[j]     = qr[32 + koff + j];
            aq1[8 + j] = qr[48 + koff + j];
        }
    }

    // --- phase 1: logits = Q K^T + mask*NEG_INF, into LDS ---
    const __bf16* kb_ptr = kws + (size_t)bh * SEQT * DEPTH;
    for (int t = 0; t < 32; ++t) {         // 8 waves * 512 keys each
        const int n0 = wave * 512 + t * 16;
        v16bf bk0, bk1;
        const __bf16* kr = kb_ptr + (size_t)(n0 + l16) * DEPTH;
        __builtin_prefetch(kr + 16 * DEPTH, 0, 0);   // next key tile
        v8f acc = {};
#pragma unroll
        for (int j = 0; j < 8; ++j) {
            bk0[j]     = kr[koff + j];
            bk0[8 + j] = kr[16 + koff + j];
            bk1[j]     = kr[32 + koff + j];
            bk1[8 + j] = kr[48 + koff + j];
        }
        acc = wmma_bf16(aq0, bk0, acc);
        acc = wmma_bf16(aq1, bk1, acc);
        const int keycol = n0 + l16;
#pragma unroll
        for (int g = 0; g < 8; ++g) {
            const int qr_ = g + half * 8;
            float lv = acc[g] + mask[(size_t)(q0 + qr_) * SEQT + keycol] * NEG_INF;
            rowbuf[qr_ * SEQT + keycol] = lv;
        }
    }
    __syncthreads();

    // --- phase 2: row softmax (16 rows x 16 threads, 256-wide chunks) ---
    {
        const int r = tid >> 4, seg = tid & 15;
        float* row = rowbuf + r * SEQT + seg * 256;
        float lmax = -3.4e38f;
        for (int j = 0; j < 256; ++j) lmax = fmaxf(lmax, row[j]);
        red[r * 16 + seg] = lmax;
        __syncthreads();
        float rmax = -3.4e38f;
        for (int i = 0; i < 16; ++i) rmax = fmaxf(rmax, red[r * 16 + i]);
        float lsum = 0.f;
        for (int j = 0; j < 256; ++j) {
            float e = __expf(row[j] - rmax);
            row[j] = e;
            lsum += e;
        }
        red[256 + r * 16 + seg] = lsum;
        __syncthreads();
        float rsum = 0.f;
        for (int i = 0; i < 16; ++i) rsum += red[256 + r * 16 + i];
        const float inv = 1.0f / rsum;
        for (int j = 0; j < 256; ++j) row[j] *= inv;
    }
#pragma unroll
    for (int i = 0; i < 4; ++i) saacc[tid + i * 256] = 0.f;
    __syncthreads();

    // --- phase 3: stream attn tile to HBM; write-once -> NT stores so the
    //     1 GB stream does not evict mask/K/V/W from the 192 MB L2 ---
    {
        float* dst = attn_out + ((size_t)bh * SEQQ + q0) * SEQT;
        for (int i = tid; i < 16 * SEQT; i += 256)
            __builtin_nontemporal_store(rowbuf[i], dst + i);
    }

    // --- phase 4: SA partial = attn(16 x 512-slab) @ V(slab x 64) ---
    const __bf16* vb = vws + (size_t)bh * SEQT * DEPTH;
    v8f pacc[4] = {{}, {}, {}, {}};
    for (int t = 0; t < 16; ++t) {         // 512 keys / 32 per WMMA
        const int kb = wave * 512 + t * 32;
        v16bf ap;
#pragma unroll
        for (int j = 0; j < 8; ++j) {
            ap[j]     = (__bf16)rowbuf[l16 * SEQT + kb + koff + j];
            ap[8 + j] = (__bf16)rowbuf[l16 * SEQT + kb + 16 + koff + j];
        }
#pragma unroll
        for (int nt = 0; nt < 4; ++nt) {
            v16bf bv;
            const __bf16* vr = vb + (size_t)kb * DEPTH + nt * 16 + l16;
#pragma unroll
            for (int j = 0; j < 8; ++j) {
                bv[j]     = vr[(size_t)(koff + j) * DEPTH];
                bv[8 + j] = vr[(size_t)(16 + koff + j) * DEPTH];
            }
            pacc[nt] = wmma_bf16(ap, bv, pacc[nt]);
        }
    }
#pragma unroll
    for (int nt = 0; nt < 4; ++nt)
#pragma unroll
        for (int g = 0; g < 8; ++g)
            atomicAdd(&saacc[(g + half * 8) * 64 + nt * 16 + l16], pacc[nt][g]);
    __syncthreads();

    // --- phase 5: SA tile -> bf16 ws in (b, s, h*64+d) layout ---
    {
        const int b = bh >> 4, h = bh & 15;
        for (int i = tid; i < 16 * 64; i += 256) {
            const int r = i >> 6, d = i & 63;
            saws[((size_t)b * SEQQ + q0 + r) * D_MODEL + h * 64 + d] =
                (__bf16)saacc[i];
        }
    }
}

// ---------------------------------------------------------------------------
// Kernel D: out = SA @ W2 + b2  (M=4096, K=1024, N=1024)
// One wave per 64x16 tile (4 sub-tiles reuse one W2 B-frag).
// ---------------------------------------------------------------------------
__global__ __launch_bounds__(32)
void proj_kernel(const __bf16* __restrict__ saws, const float* __restrict__ W2,
                 const float* __restrict__ b2, float* __restrict__ out)
{
    const int lane = threadIdx.x & 31;
    const int half = lane >> 4;
    const int l16  = lane & 15;
    const int koff = half * 8;
    const int n0 = blockIdx.x * 16;
    const int m0 = blockIdx.y * 64;

    v8f acc[4] = {{}, {}, {}, {}};
    for (int kk = 0; kk < D_MODEL; kk += 32) {
        v16bf b;
        const float* wr = W2 + (size_t)(kk + koff) * D_MODEL + n0 + l16;
#pragma unroll
        for (int j = 0; j < 8; ++j) {
            b[j]     = (__bf16)wr[(size_t)j * D_MODEL];
            b[8 + j] = (__bf16)wr[(size_t)(16 + j) * D_MODEL];
        }
#pragma unroll
        for (int sm = 0; sm < 4; ++sm) {
            v16bf a;
            const __bf16* ar = saws + (size_t)(m0 + sm * 16 + l16) * D_MODEL + kk + koff;
#pragma unroll
            for (int j = 0; j < 8; ++j) {
                a[j]     = ar[j];
                a[8 + j] = ar[16 + j];
            }
            acc[sm] = wmma_bf16(a, b, acc[sm]);
        }
    }
    const float bias = b2[n0 + l16];
#pragma unroll
    for (int sm = 0; sm < 4; ++sm)
#pragma unroll
        for (int g = 0; g < 8; ++g) {
            const int m = m0 + sm * 16 + g + half * 8;
            __builtin_nontemporal_store(acc[sm][g] + bias,
                                        out + (size_t)m * D_MODEL + n0 + l16);
        }
}

// ---------------------------------------------------------------------------
extern "C" void kernel_launch(void* const* d_in, const int* in_sizes, int n_in,
                              void* d_out, int out_size, void* d_ws, size_t ws_size,
                              hipStream_t stream)
{
    const float* X    = (const float*)d_in[0];
    const float* past = (const float*)d_in[1];
    const float* mask = (const float*)d_in[2];
    const float* W1   = (const float*)d_in[3];
    const float* b1   = (const float*)d_in[4];
    const float* W2   = (const float*)d_in[5];
    const float* b2   = (const float*)d_in[6];

    float* out     = (float*)d_out;                                       // 4,194,304
    float* present = out + (size_t)M_TOT * D_MODEL;                       // 8,388,608
    float* attn    = present + (size_t)BATCH * 2 * NHEAD * SEQQ * DEPTH;  // 268,435,456

    __bf16* qws  = (__bf16*)d_ws;                                  //  8 MB
    __bf16* kws  = qws + (size_t)BATCH * NHEAD * SEQQ * DEPTH;     // 16 MB
    __bf16* vws  = kws + (size_t)BATCH * NHEAD * SEQT * DEPTH;     // 16 MB
    __bf16* saws = vws + (size_t)BATCH * NHEAD * SEQT * DEPTH;     //  8 MB

    qkv_kernel<<<dim3(3 * D_MODEL / 16, M_TOT / 64), 32, 0, stream>>>(
        X, W1, b1, present, qws, kws, vws);

    past_kernel<<<(BATCH * NHEAD * SEQP * DEPTH) / 256, 256, 0, stream>>>(
        past, kws, vws);

    const size_t shmem = (size_t)(16 * SEQT + 512 + 16 * 64) * sizeof(float)
                         + (size_t)(16 * DEPTH) * sizeof(__bf16);
    attn_kernel<<<dim3(SEQQ / 16, BATCH * NHEAD), 256, shmem, stream>>>(
        qws, kws, vws, mask, attn, saws);

    proj_kernel<<<dim3(D_MODEL / 16, M_TOT / 64), 32, 0, stream>>>(
        saws, W2, b2, out);
}
